// MultiHeadedAttention_42348377538721
// MI455X (gfx1250) — matrix operand order repaired and checked by reading, not compiled, not run on hardware
//
#include <hip/hip_runtime.h>

// ---------------------------------------------------------------------------
// MultiHeadedAttention for MI455X (gfx1250, wave32, WMMA bf16 16x16x32 + TDM)
// B=8, L=1024, D=1024, H=16, dk=64
// ---------------------------------------------------------------------------

typedef __bf16 bf16_t;
typedef __attribute__((ext_vector_type(4)))  __bf16 v4bf;
typedef __attribute__((ext_vector_type(8)))  __bf16 v8bf;
typedef __attribute__((ext_vector_type(16))) __bf16 v16bf;
typedef __attribute__((ext_vector_type(8)))  float  v8f;
typedef __attribute__((ext_vector_type(4)))  float  f32x4;
typedef __attribute__((ext_vector_type(4)))  unsigned int u32x4;
typedef __attribute__((ext_vector_type(8)))  int    i32x8;
typedef __attribute__((ext_vector_type(4)))  int    i32x4;

static constexpr int Bb = 8;
static constexpr int Ll = 1024;
static constexpr int Dd = 1024;
static constexpr int Hh = 16;
static constexpr int DK = 64;
static constexpr int Mm = Bb * Ll;       // 8192
static constexpr int BH = Bb * Hh;       // 128

__device__ __forceinline__ bf16_t f2bf(float f) {
    union { float f; unsigned int u; } c; c.f = f;
    unsigned int r = c.u + 0x7FFFu + ((c.u >> 16) & 1u);
    union { unsigned short s; bf16_t b; } o; o.s = (unsigned short)(r >> 16);
    return o.b;
}

__device__ __forceinline__ v16bf frag_cat(const bf16_t* p0, const bf16_t* p1) {
    v8bf lo = *(const v8bf*)p0;
    v8bf hi = *(const v8bf*)p1;
    return __builtin_shufflevector(lo, hi, 0,1,2,3,4,5,6,7,8,9,10,11,12,13,14,15);
}

__device__ __forceinline__ v8f wmma_bf16(v16bf a, v16bf b, v8f c) {
    return __builtin_amdgcn_wmma_f32_16x16x32_bf16(
        false, a, false, b, (short)0, c, false, false);
}

__device__ __forceinline__ v8f zero8() {
    v8f z;
#pragma unroll
    for (int e = 0; e < 8; ++e) z[e] = 0.0f;
    return z;
}

// ---------------------------------------------------------------------------
// Tensor Data Mover: build a D# (groups 0/1 per CDNA5 ISA)
// and issue TENSOR_LOAD_TO_LDS (6-arg clang-23 form).  2D tile (tile_d1 rows
// of tile_d0 elements of 2 bytes, row stride stride0 elements); tile_d1==0 ->
// 1D.  LDS padding of pad_amount(code) DWORDs every 2^pad_interval qwords
// de-conflicts LDS banks.
// ---------------------------------------------------------------------------
__device__ __forceinline__ void tdm_load_2d(unsigned lds_off, const void* gptr,
                                            unsigned tile_d0, unsigned tile_d1,
                                            unsigned long long stride0,
                                            unsigned pad_interval, unsigned pad_amount) {
    const unsigned long long g = (unsigned long long)gptr;
    u32x4 g0;
    g0[0] = 1u;                                               // count=1, no gather
    g0[1] = lds_off;                                          // lds_addr (bytes)
    g0[2] = (unsigned)g;                                      // global_addr[31:0]
    g0[3] = (unsigned)((g >> 32) & 0x01FFFFFFu) | (2u << 30); // addr[56:32] | type=2

    const unsigned td0 = 1u << 20;     // generous tensor dims: no OOB clipping
    const unsigned td1 = 1u << 20;
    i32x8 g1;
    g1[0] = (int)((1u << 16) |                       // data_size = 2 bytes
                  (1u << 20) |                       // pad_enable
                  (pad_interval << 22) |
                  (pad_amount << 25));
    g1[1] = (int)((td0 & 0xFFFFu) << 16);                            // tdim0 lo
    g1[2] = (int)(((td0 >> 16) & 0xFFFFu) | ((td1 & 0xFFFFu) << 16));// tdim0 hi|tdim1 lo
    g1[3] = (int)(((td1 >> 16) & 0xFFFFu) | (tile_d0 << 16));        // tdim1 hi|tile0
    g1[4] = (int)(tile_d1 & 0xFFFFu);                                // tile1, tile2=0
    g1[5] = (int)(unsigned)(stride0 & 0xFFFFFFFFull);                // stride0[31:0]
    g1[6] = (int)(unsigned)((stride0 >> 32) & 0xFFFFull);            // stride0[47:32]
    g1[7] = 0;                                                       // stride1 unused

    i32x4 z4;
#pragma unroll
    for (int e = 0; e < 4; ++e) z4[e] = 0;
    i32x8 z8;
#pragma unroll
    for (int e = 0; e < 8; ++e) z8[e] = 0;
    __builtin_amdgcn_tensor_load_to_lds(g0, g1, z4, z4, z8, 0);
}

// ---------------------------------------------------------------------------
// Kernel 1: Y = X @ W^T  (M=8192, N=1024, K=1024), f32 in -> bf16 out.
// Block tile 128x128, 256 threads = 8 waves (2x4), wave tile 64x32 (4x2 WMMA).
// transpose_out==0: Y[b][h][l][d]  (Q,K) ; ==1: Y[b][h][d][l]  (V^T)
// ---------------------------------------------------------------------------
static constexpr int LDS_STRIDE = 40;  // 32 + 8 halfs padding (80B rows)

__global__ __launch_bounds__(256)
void proj_kernel(const float* __restrict__ X, const float* __restrict__ W,
                 bf16_t* __restrict__ Y, int transpose_out) {
    __shared__ __align__(16) bf16_t As[128 * LDS_STRIDE];
    __shared__ __align__(16) bf16_t Bs[128 * LDS_STRIDE];

    const int tid  = threadIdx.x;
    const int lane = tid & 31;
    const int wave = tid >> 5;
    const int m0   = blockIdx.y * 128;
    const int n0   = blockIdx.x * 128;
    const int wm   = wave >> 2;
    const int wn   = wave & 3;
    const int lr    = lane & 15;
    const int lhalf = lane >> 4;

    const int lrow = tid >> 3;
    const int lcol = (tid & 7) * 4;

    v8f acc[4][2];
#pragma unroll
    for (int i = 0; i < 4; ++i)
#pragma unroll
        for (int j = 0; j < 2; ++j) acc[i][j] = zero8();

    for (int kt = 0; kt < Dd; kt += 32) {
        __syncthreads();
#pragma unroll
        for (int i = 0; i < 4; ++i) {
            const int row = lrow + i * 32;
            f32x4 xa = *(const f32x4*)(X + (size_t)(m0 + row) * Dd + kt + lcol);
            f32x4 wa = *(const f32x4*)(W + (size_t)(n0 + row) * Dd + kt + lcol);
            v4bf xp, wp;
#pragma unroll
            for (int j = 0; j < 4; ++j) { xp[j] = f2bf(xa[j]); wp[j] = f2bf(wa[j]); }
            *(v4bf*)&As[row * LDS_STRIDE + lcol] = xp;
            *(v4bf*)&Bs[row * LDS_STRIDE + lcol] = wp;
        }
        __syncthreads();

        v16bf afrag[4];
#pragma unroll
        for (int mi = 0; mi < 4; ++mi) {
            const bf16_t* base = &As[(wm * 64 + mi * 16 + lr) * LDS_STRIDE + lhalf * 8];
            afrag[mi] = frag_cat(base, base + 16);
        }
        v16bf bfrag[2];
#pragma unroll
        for (int ni = 0; ni < 2; ++ni) {
            const bf16_t* base = &Bs[(wn * 32 + ni * 16 + lr) * LDS_STRIDE + lhalf * 8];
            bfrag[ni] = frag_cat(base, base + 16);
        }
#pragma unroll
        for (int mi = 0; mi < 4; ++mi)
#pragma unroll
            for (int ni = 0; ni < 2; ++ni)
                acc[mi][ni] = wmma_bf16(afrag[mi], bfrag[ni], acc[mi][ni]);
    }

    const int mbase = m0 + wm * 64 + lhalf * 8;
#pragma unroll
    for (int mi = 0; mi < 4; ++mi) {
#pragma unroll
        for (int ni = 0; ni < 2; ++ni) {
            const int n = n0 + wn * 32 + ni * 16 + lr;
            const int h = n >> 6, d = n & 63;
#pragma unroll
            for (int r = 0; r < 8; ++r) {
                const int m = mbase + mi * 16 + r;
                const int b = m >> 10, l = m & 1023;
                size_t idx = transpose_out
                    ? (((size_t)(b * Hh + h) * DK + d) * Ll + l)
                    : (((size_t)(b * Hh + h) * Ll + l) * DK + d);
                Y[idx] = f2bf(acc[mi][ni][r]);
            }
        }
    }
}

// ---------------------------------------------------------------------------
// Kernel 2: raw scores S = Q K^T / 8, mask applied, written to d_out.
// TDM double-buffers 128-key chunks of K into LDS (16KB -> 18KB padded):
// wave 0 waits TENSORcnt, barrier, issues next chunk, everyone computes.
// LDS row stride 144B (pad 4 DWORDs / 32 DWORDs) => conflict-free b128 reads.
// ---------------------------------------------------------------------------
static constexpr int KROW = 72;              // halfs per padded K row (144B)
static constexpr int KCHUNK_BYTES = 128 * KROW * 2;   // 18KB per buffer

__global__ __launch_bounds__(256)
void scores_kernel(const bf16_t* __restrict__ Q, const bf16_t* __restrict__ K,
                   const unsigned char* __restrict__ mask, float* __restrict__ S) {
    __shared__ __align__(16) bf16_t ldsK[2 * 128 * KROW];

    const int tid  = threadIdx.x;
    const int lane = tid & 31;
    const int wave = tid >> 5;
    const int bh   = blockIdx.y;
    const int b    = bh >> 4;
    const int q0   = blockIdx.x * 128 + wave * 16;
    const int lr    = lane & 15;
    const int lhalf = lane >> 4;

    const bf16_t* Qb = Q + (size_t)bh * Ll * DK;
    const bf16_t* Kb = K + (size_t)bh * Ll * DK;
    const unsigned lds_base = (unsigned)(uintptr_t)(void*)ldsK;

    const bf16_t* qr = Qb + (size_t)(q0 + lr) * DK + lhalf * 8;
    const v16bf a0 = frag_cat(qr, qr + 16);        // d 0..31
    const v16bf a1 = frag_cat(qr + 32, qr + 48);   // d 32..63

    // Prologue: chunk 0 (128 K-rows, contiguous 16KB, 1D tile of 8192 halfs;
    // pad 4 DWORDs every 32 DWORDs -> +16B after each 128B row).
    if (wave == 0)
        tdm_load_2d(lds_base, Kb, 8192u, 0u, 8192ull, 4u, 3u);

    const int qrow0 = q0 + lhalf * 8;
    for (int ch = 0; ch < 8; ++ch) {
        if (wave == 0) __builtin_amdgcn_s_wait_tensorcnt(0);
        __syncthreads();
        if (wave == 0 && ch + 1 < 8)
            tdm_load_2d(lds_base + (unsigned)(((ch + 1) & 1) * KCHUNK_BYTES),
                        Kb + (size_t)(ch + 1) * 128 * DK, 8192u, 0u, 8192ull, 4u, 3u);

        const bf16_t* kbuf = ldsK + (ch & 1) * (128 * KROW);
#pragma unroll
        for (int t = 0; t < 8; ++t) {
            const bf16_t* p = kbuf + (t * 16 + lr) * KROW + lhalf * 8;
            const v16bf b0 = frag_cat(p, p + 16);         // d 0..31
            const v16bf b1 = frag_cat(p + 32, p + 48);    // d 32..63
            v8f acc = zero8();
            acc = wmma_bf16(a0, b0, acc);
            acc = wmma_bf16(a1, b1, acc);

            const int kcol = ch * 128 + t * 16 + lr;
#pragma unroll
            for (int r = 0; r < 8; ++r) {
                float s = acc[r] * 0.125f;   // 1/sqrt(64)
                const unsigned char mk =
                    mask[((size_t)b * Ll + (qrow0 + r)) * Ll + kcol];
                if (mk) s = -__builtin_inff();
                S[((size_t)bh * Ll + (qrow0 + r)) * Ll + kcol] = s;
            }
        }
    }
}

// ---------------------------------------------------------------------------
// Kernel 3: per-row softmax stats (max, sum exp). One wave per row of 1024.
// ---------------------------------------------------------------------------
__global__ __launch_bounds__(256)
void stats_kernel(const float* __restrict__ S,
                  float* __restrict__ Mv, float* __restrict__ Lv) {
    const int row  = blockIdx.x * 8 + (threadIdx.x >> 5);
    const int lane = threadIdx.x & 31;
    const float* p = S + (size_t)row * Ll;

    f32x4 v[8];
    float mx = -__builtin_inff();
#pragma unroll
    for (int i = 0; i < 8; ++i) {
        v[i] = *(const f32x4*)(p + lane * 4 + i * 128);
#pragma unroll
        for (int j = 0; j < 4; ++j) mx = fmaxf(mx, v[i][j]);
    }
#pragma unroll
    for (int off = 16; off > 0; off >>= 1) mx = fmaxf(mx, __shfl_xor(mx, off, 32));

    float sum = 0.0f;
#pragma unroll
    for (int i = 0; i < 8; ++i)
#pragma unroll
        for (int j = 0; j < 4; ++j) sum += __expf(v[i][j] - mx);
#pragma unroll
    for (int off = 16; off > 0; off >>= 1) sum += __shfl_xor(sum, off, 32);

    if (lane == 0) { Mv[row] = mx; Lv[row] = sum; }
}

// ---------------------------------------------------------------------------
// Kernel 4: fused probs rewrite + P@V.  V^T 64x32 tiles are TDM-prefetched
// into LDS (pad 4 DWORDs / 16 DWORDs -> 80B row stride, conflict-free) and
// shared by all 8 waves; each lane rewrites exactly its A-fragment's S
// elements as probabilities (attn output) and feeds WMMA.
// ---------------------------------------------------------------------------
static constexpr int VROW = 40;                       // halfs per padded V row (80B)
static constexpr int VCHUNK_BYTES = 64 * VROW * 2;    // 5KB per buffer

__global__ __launch_bounds__(256)
void pv_kernel(float* __restrict__ S, const bf16_t* __restrict__ VT,
               const float* __restrict__ Mv, const float* __restrict__ Lv,
               float* __restrict__ O) {
    __shared__ __align__(16) bf16_t ldsV[2 * 64 * VROW];

    const int tid  = threadIdx.x;
    const int lane = tid & 31;
    const int wave = tid >> 5;
    const int bh   = blockIdx.y;
    const int q0   = blockIdx.x * 128 + wave * 16;
    const int lr    = lane & 15;
    const int lhalf = lane >> 4;

    const int arow = q0 + lr;
    const float m    = Mv[(size_t)bh * Ll + arow];
    const float invl = 1.0f / Lv[(size_t)bh * Ll + arow];

    float* Srow = S + ((size_t)bh * Ll + arow) * Ll;
    const bf16_t* VTb = VT + (size_t)bh * DK * Ll;
    const unsigned lds_base = (unsigned)(uintptr_t)(void*)ldsV;

    v8f acc[4];
#pragma unroll
    for (int ni = 0; ni < 4; ++ni) acc[ni] = zero8();

    // Prologue: V^T tile for kt=0 (64 rows x 32 halfs, row stride 1024 halfs).
    if (wave == 0)
        tdm_load_2d(lds_base, VTb, 32u, 64u, 1024ull, 3u, 3u);

    for (int c = 0; c < 32; ++c) {
        if (wave == 0) __builtin_amdgcn_s_wait_tensorcnt(0);
        __syncthreads();
        if (wave == 0 && c + 1 < 32)
            tdm_load_2d(lds_base + (unsigned)(((c + 1) & 1) * VCHUNK_BYTES),
                        VTb + (size_t)(c + 1) * 32, 32u, 64u, 1024ull, 3u, 3u);

        const int kt = c * 32;
        const int cbase = kt + lhalf * 8;
        f32x4 s0 = *(const f32x4*)(Srow + cbase);
        f32x4 s1 = *(const f32x4*)(Srow + cbase + 4);
        f32x4 s2 = *(const f32x4*)(Srow + cbase + 16);
        f32x4 s3 = *(const f32x4*)(Srow + cbase + 20);
#pragma unroll
        for (int j = 0; j < 4; ++j) {
            s0[j] = __expf(s0[j] - m) * invl;
            s1[j] = __expf(s1[j] - m) * invl;
            s2[j] = __expf(s2[j] - m) * invl;
            s3[j] = __expf(s3[j] - m) * invl;
        }
        *(f32x4*)(Srow + cbase)      = s0;   // attn probabilities out (in place)
        *(f32x4*)(Srow + cbase + 4)  = s1;
        *(f32x4*)(Srow + cbase + 16) = s2;
        *(f32x4*)(Srow + cbase + 20) = s3;

        v16bf af;
#pragma unroll
        for (int j = 0; j < 4; ++j) {
            af[j]      = f2bf(s0[j]);
            af[4 + j]  = f2bf(s1[j]);
            af[8 + j]  = f2bf(s2[j]);
            af[12 + j] = f2bf(s3[j]);
        }
        const bf16_t* vbuf = ldsV + (c & 1) * (64 * VROW);
#pragma unroll
        for (int ni = 0; ni < 4; ++ni) {
            const bf16_t* p = vbuf + (ni * 16 + lr) * VROW + lhalf * 8;
            const v16bf bf_ = frag_cat(p, p + 16);
            acc[ni] = wmma_bf16(af, bf_, acc[ni]);
        }
    }

    const int h = bh & 15, b = bh >> 4;
#pragma unroll
    for (int ni = 0; ni < 4; ++ni) {
        const int d = h * 64 + ni * 16 + lr;
#pragma unroll
        for (int r = 0; r < 8; ++r) {
            const int l = q0 + lhalf * 8 + r;
            O[((size_t)b * Ll + l) * Dd + d] = acc[ni][r];
        }
    }
}

// ---------------------------------------------------------------------------
// Kernel 5: residual add + LayerNorm over D=1024. One 256-thread WG per row.
// ---------------------------------------------------------------------------
__global__ __launch_bounds__(256)
void ln_kernel(const float* __restrict__ O, const float* __restrict__ query,
               const float* __restrict__ gamma, const float* __restrict__ beta,
               float* __restrict__ out) {
    __shared__ float red1[8], red2[8];
    const int row = blockIdx.x;
    const int tid = threadIdx.x;
    const int lane = tid & 31;
    const int wave = tid >> 5;

    f32x4 r = *(const f32x4*)(O + (size_t)row * Dd + tid * 4)
            + *(const f32x4*)(query + (size_t)row * Dd + tid * 4);

    float s1 = r[0] + r[1] + r[2] + r[3];
    float s2 = r[0]*r[0] + r[1]*r[1] + r[2]*r[2] + r[3]*r[3];
#pragma unroll
    for (int off = 16; off > 0; off >>= 1) {
        s1 += __shfl_xor(s1, off, 32);
        s2 += __shfl_xor(s2, off, 32);
    }
    if (lane == 0) { red1[wave] = s1; red2[wave] = s2; }
    __syncthreads();
    float t1 = 0.0f, t2 = 0.0f;
#pragma unroll
    for (int w = 0; w < 8; ++w) { t1 += red1[w]; t2 += red2[w]; }

    const float mu   = t1 * (1.0f / Dd);
    const float var  = t2 * (1.0f / Dd) - mu * mu;
    const float rstd = rsqrtf(var + 1e-5f);

    f32x4 g  = *(const f32x4*)(gamma + tid * 4);
    f32x4 be = *(const f32x4*)(beta + tid * 4);
    f32x4 o;
#pragma unroll
    for (int j = 0; j < 4; ++j) o[j] = (r[j] - mu) * rstd * g[j] + be[j];
    *(f32x4*)(out + (size_t)row * Dd + tid * 4) = o;
}

// ---------------------------------------------------------------------------
extern "C" void kernel_launch(void* const* d_in, const int* in_sizes, int n_in,
                              void* d_out, int out_size, void* d_ws, size_t ws_size,
                              hipStream_t stream) {
    const float*         key   = (const float*)d_in[0];
    const float*         value = (const float*)d_in[1];
    const float*         query = (const float*)d_in[2];
    const unsigned char* mask  = (const unsigned char*)d_in[3];
    const float*         Wq    = (const float*)d_in[4];
    const float*         Wk    = (const float*)d_in[5];
    const float*         Wv    = (const float*)d_in[6];
    const float*         gamma = (const float*)d_in[7];
    const float*         beta  = (const float*)d_in[8];

    float* ln_out = (float*)d_out;                               // B*L*D
    float* attn   = (float*)d_out + (size_t)Bb * Ll * Dd;        // B*H*L*L

    char* ws = (char*)d_ws;
    size_t off = 0;
    bf16_t* qw = (bf16_t*)(ws + off); off += (size_t)Bb * Hh * Ll * DK * sizeof(bf16_t);
    bf16_t* kw = (bf16_t*)(ws + off); off += (size_t)Bb * Hh * Ll * DK * sizeof(bf16_t);
    bf16_t* vt = (bf16_t*)(ws + off); off += (size_t)Bb * Hh * DK * Ll * sizeof(bf16_t);
    float*  Mv = (float*)(ws + off);  off += (size_t)BH * Ll * sizeof(float);
    float*  Lv = (float*)(ws + off);  off += (size_t)BH * Ll * sizeof(float);
    float*  Ob = (float*)(ws + off);  off += (size_t)Bb * Ll * Dd * sizeof(float);
    (void)ws_size; (void)in_sizes; (void)n_in; (void)out_size;

    const dim3 blk(256);
    proj_kernel<<<dim3(Dd / 128, Mm / 128), blk, 0, stream>>>(query, Wq, qw, 0);
    proj_kernel<<<dim3(Dd / 128, Mm / 128), blk, 0, stream>>>(key,   Wk, kw, 0);
    proj_kernel<<<dim3(Dd / 128, Mm / 128), blk, 0, stream>>>(value, Wv, vt, 1);
    scores_kernel<<<dim3(Ll / 128, BH), blk, 0, stream>>>(qw, kw, mask, attn);
    stats_kernel<<<dim3(BH * Ll / 8), blk, 0, stream>>>(attn, Mv, Lv);
    pv_kernel<<<dim3(Ll / 128, BH), blk, 0, stream>>>(attn, vt, Mv, Lv, Ob);
    ln_kernel<<<dim3(Bb * Ll), blk, 0, stream>>>(Ob, query, gamma, beta, ln_out);
}
